// Attention_52836687676078
// MI455X (gfx1250) — matrix operand order repaired and checked by reading, not compiled, not run on hardware
//
#include <hip/hip_runtime.h>

typedef _Float16 v16h __attribute__((ext_vector_type(16)));
typedef _Float16 v8h  __attribute__((ext_vector_type(8)));
typedef _Float16 v4h  __attribute__((ext_vector_type(4)));
typedef float    v8f  __attribute__((ext_vector_type(8)));

#define S_LEN   4096
#define DHEAD   64
#define NHEAD   16
#define QTILE   128          // query rows per block (8 waves x 16 rows)
#define KTILE   64           // keys per LDS stage
#define LSTR    72           // LDS row stride in halves (144B = 9*16B, keeps b128 aligned)
#define NEG_BIG (-3.0e38f)

static __device__ __forceinline__ v16h cat8(v8h a, v8h b) {
    return __builtin_shufflevector(a, b, 0,1,2,3,4,5,6,7,8,9,10,11,12,13,14,15);
}

__global__ __launch_bounds__(256) void fa_causal_gfx1250(
    const float* __restrict__ q, const float* __restrict__ k,
    const float* __restrict__ v, float* __restrict__ out)
{
    // LDS: K tile [key][d], V tile transposed [d][key], P scratch per wave [query][key]
    __shared__ _Float16 Kt[KTILE * LSTR];
    __shared__ _Float16 Vt[DHEAD * LSTR];
    __shared__ _Float16 Pt[QTILE * LSTR];

    const int h   = blockIdx.x / (S_LEN / QTILE);
    const int qt  = blockIdx.x % (S_LEN / QTILE);
    const int q0  = qt * QTILE;

    const int tid    = threadIdx.x;
    const int wave   = tid >> 5;
    const int lane   = tid & 31;
    const int lane16 = lane & 15;
    const int hi     = lane >> 4;

    const int wave_q0 = q0 + wave * 16;
    const int qrow    = wave_q0 + lane16;          // this lane's query row (N index)

    const float qscale = 0.125f * 1.44269504088896f;   // 1/sqrt(64) * log2(e)

    // ---- Q B-fragments (Q^T: contract = d, N = query), built once ----
    // B layout: VGPR j = contract pair (2j,2j+1) + 16*hi  ->  16 contiguous d values
    v16h qb[2];
    {
        const float* qp = q + ((size_t)h * S_LEN + qrow) * DHEAD;
        #pragma unroll
        for (int c = 0; c < 2; ++c) {
            const float4* p4 = (const float4*)(qp + c * 32 + hi * 16);
            float tf[16];
            #pragma unroll
            for (int i = 0; i < 4; ++i) {
                float4 f = p4[i];
                tf[i*4+0] = f.x; tf[i*4+1] = f.y; tf[i*4+2] = f.z; tf[i*4+3] = f.w;
            }
            v16h b;
            #pragma unroll
            for (int j = 0; j < 16; ++j) b[j] = (_Float16)(tf[j] * qscale);
            qb[c] = b;
        }
    }

    // Online-softmax state (per lane == per query row)
    float m_run = NEG_BIG;
    float l_run = 0.0f;
    v8f acc_o[4];                                  // O^T frags: M = d chunk, N = query
    #pragma unroll
    for (int nd = 0; nd < 4; ++nd) acc_o[nd] = (v8f){0,0,0,0,0,0,0,0};

    // K fill assignment: thread -> one key row, 16 d values (coalesced float4 x4)
    const int f_key = tid >> 2;                    // 0..63
    const int f_d0  = (tid & 3) * 16;              // 0,16,32,48
    // V fill assignment: thread -> 4x4 block (4 keys x 4 d), register transpose
    const int vk0   = (tid >> 4) * 4;              // 0..60 step 4 (key base)
    const int vd0   = (tid & 15) * 4;              // 0..60 step 4 (d base)

    const int kv_end = q0 + QTILE - KTILE;         // last tile any wave needs

    // Software-pipelined global tile loads (registers hold tile i+1 during compute of i)
    float4 kreg[4], vreg[4];
    auto load_tile = [&](int kv0) {
        const float* kp = k + ((size_t)h * S_LEN + kv0 + f_key) * DHEAD + f_d0;
        #pragma unroll
        for (int i = 0; i < 4; ++i) kreg[i] = ((const float4*)kp)[i];
        const float* vp = v + ((size_t)h * S_LEN + kv0 + vk0) * DHEAD + vd0;
        #pragma unroll
        for (int r = 0; r < 4; ++r) vreg[r] = *(const float4*)(vp + r * DHEAD);
        if (kv0 + KTILE <= kv_end) {               // L2 prefetch for tile after next
            __builtin_prefetch(kp + (size_t)KTILE * DHEAD, 0, 3);
            __builtin_prefetch(vp + (size_t)KTILE * DHEAD, 0, 3);
        }
    };
    auto store_tile = [&]() {
        #pragma unroll
        for (int i = 0; i < 4; ++i) {
            float4 f = kreg[i];
            v4h hv = { (_Float16)f.x, (_Float16)f.y, (_Float16)f.z, (_Float16)f.w };
            *(v4h*)&Kt[f_key * LSTR + f_d0 + i * 4] = hv;            // ds_store_b64
        }
        float a[4][4];
        #pragma unroll
        for (int r = 0; r < 4; ++r) {
            a[r][0] = vreg[r].x; a[r][1] = vreg[r].y;
            a[r][2] = vreg[r].z; a[r][3] = vreg[r].w;
        }
        #pragma unroll
        for (int c = 0; c < 4; ++c) {              // transposed: 4 consecutive keys
            v4h col = { (_Float16)a[0][c], (_Float16)a[1][c],
                        (_Float16)a[2][c], (_Float16)a[3][c] };
            *(v4h*)&Vt[(vd0 + c) * LSTR + vk0] = col;                // ds_store_b64
        }
    };

    load_tile(0);

    for (int kv0 = 0; kv0 <= kv_end; kv0 += KTILE) {
        store_tile();                               // tile i regs -> LDS (f16)
        __syncthreads();
        if (kv0 + KTILE <= kv_end) load_tile(kv0 + KTILE);   // tile i+1 in flight

        // Wave-uniform causal skip (keeps EXEC all-ones around WMMA)
        if (kv0 <= wave_q0 + 15) {
            // ---- S^T = K(16key x 32d) x Q^T(32d x 16query), 4 key chunks ----
            v8f st[4];
            #pragma unroll
            for (int n = 0; n < 4; ++n) {
                v8f acc = (v8f){0,0,0,0,0,0,0,0};
                #pragma unroll
                for (int c = 0; c < 2; ++c) {
                    // A layout: row = key (lane16), 16 contiguous cols at 32c+16hi
                    const _Float16* kr = &Kt[(n*16 + lane16) * LSTR + c*32 + hi*16];
                    v16h a = cat8(*(const v8h*)kr, *(const v8h*)(kr + 8));
                    acc = __builtin_amdgcn_wmma_f32_16x16x32_f16(
                              false, a, false, qb[c], (short)0, acc, false, false);
                }
                st[n] = acc;
            }

            // ---- causal mask (diagonal tiles only) ----
            if (kv0 + KTILE - 1 > wave_q0) {
                #pragma unroll
                for (int n = 0; n < 4; ++n)
                    #pragma unroll
                    for (int g = 0; g < 8; ++g) {
                        int key = kv0 + n*16 + g + 8*hi;
                        if (key > qrow) st[n][g] = NEG_BIG;
                    }
            }

            // ---- online softmax (log2 domain) ----
            float tmax = NEG_BIG;
            #pragma unroll
            for (int n = 0; n < 4; ++n)
                #pragma unroll
                for (int g = 0; g < 8; ++g) tmax = fmaxf(tmax, st[n][g]);
            tmax = fmaxf(tmax, __shfl_xor(tmax, 16));
            const float m_new = fmaxf(m_run, tmax);
            const float alpha = exp2f(m_run - m_new);
            m_run = m_new;

            float psum = 0.0f;
            _Float16* prow = &Pt[(wave*16 + lane16) * LSTR];
            #pragma unroll
            for (int n = 0; n < 4; ++n) {
                v8h ph;
                #pragma unroll
                for (int g = 0; g < 8; ++g) {
                    float p = exp2f(st[n][g] - m_new);
                    psum += p;
                    ph[g] = (_Float16)p;
                }
                // P[query][key]: cols 16n+8hi..+7 contiguous -> one ds_store_b128
                *(v8h*)&prow[n*16 + 8*hi] = ph;
            }
            psum += __shfl_xor(psum, 16);
            l_run = l_run * alpha + psum;
            #pragma unroll
            for (int nd = 0; nd < 4; ++nd)
                #pragma unroll
                for (int g = 0; g < 8; ++g) acc_o[nd][g] *= alpha;

            // P stores -> P loads within same wave: drain DS counter
            asm volatile("s_wait_dscnt 0" ::: "memory");

            // ---- O^T += V^T(16d x 32key) x P^T(32key x 16query) ----
            #pragma unroll
            for (int s = 0; s < 2; ++s) {
                const _Float16* pr = &Pt[(wave*16 + lane16) * LSTR + s*32 + hi*8];
                v16h pb = cat8(*(const v8h*)pr, *(const v8h*)(pr + 16));
                #pragma unroll
                for (int nd = 0; nd < 4; ++nd) {
                    const _Float16* vr = &Vt[(nd*16 + lane16) * LSTR + s*32 + hi*8];
                    v16h va = cat8(*(const v8h*)vr, *(const v8h*)(vr + 16));
                    acc_o[nd] = __builtin_amdgcn_wmma_f32_16x16x32_f16(
                                    false, va, false, pb, (short)0, acc_o[nd], false, false);
                }
            }
        }
        __syncthreads();
    }

    // ---- normalize and write O: element g of acc_o[nd] = O[qrow][nd*16 + g + 8*hi] ----
    const float inv_l = 1.0f / l_run;
    float* op = out + ((size_t)h * S_LEN + qrow) * DHEAD + 8 * hi;
    #pragma unroll
    for (int nd = 0; nd < 4; ++nd) {
        float4 s0 = { acc_o[nd][0] * inv_l, acc_o[nd][1] * inv_l,
                      acc_o[nd][2] * inv_l, acc_o[nd][3] * inv_l };
        float4 s1 = { acc_o[nd][4] * inv_l, acc_o[nd][5] * inv_l,
                      acc_o[nd][6] * inv_l, acc_o[nd][7] * inv_l };
        *(float4*)(op + nd*16)     = s0;           // global_store_b128
        *(float4*)(op + nd*16 + 4) = s1;           // global_store_b128
    }
}

extern "C" void kernel_launch(void* const* d_in, const int* in_sizes, int n_in,
                              void* d_out, int out_size, void* d_ws, size_t ws_size,
                              hipStream_t stream) {
    (void)in_sizes; (void)n_in; (void)out_size; (void)d_ws; (void)ws_size;
    const float* q = (const float*)d_in[0];
    const float* k = (const float*)d_in[1];
    const float* v = (const float*)d_in[2];
    float* out = (float*)d_out;
    dim3 grid(NHEAD * (S_LEN / QTILE));
    fa_causal_gfx1250<<<grid, 256, 0, stream>>>(q, k, v, out);
}
